// RSNN_4604204941427
// MI455X (gfx1250) — matrix operand order repaired and checked by reading, not compiled, not run on hardware
//
#include <hip/hip_runtime.h>
#include <stddef.h>

typedef unsigned short ushort_t;
typedef __bf16 bf16_t;
typedef float v8f   __attribute__((ext_vector_type(8)));
typedef bf16_t v16bf __attribute__((ext_vector_type(16)));
typedef unsigned int v4u __attribute__((ext_vector_type(4)));
typedef int v4i __attribute__((ext_vector_type(4)));
typedef int v8i __attribute__((ext_vector_type(8)));

#define DECAY   0.9f
#define THRESH  0.5f
#define ALPHA_  0.9f
#define BETA_   1.8f
#define RHO_    0.98f
#define KAPPA_  0.85f

#define T_STEPS 512
#define BATCH   64
#define NIN     256
#define HID     1024
#define NOUT    128
#define FANIN   1280
#define NWG     16

// d_ws layout (bytes):
//   [0,256)                      : sync (cnt @ u32[0], gen @ u32[16])
//   [256, +16*163840)            : WB  hidden weight fragments (bf16, per-WG slices)
//   [.., +8*32768)               : WOB readout weight fragments
//   [.., +2*64*1024*2)           : SB  double-buffered spike matrix (bf16)
#define WS_WB_OFF   256
#define WS_WOB_OFF  (WS_WB_OFF + 16 * 163840)
#define WS_SB_OFF   (WS_WOB_OFF + 8 * 32768)

union BFPack {
  struct { uint4 lo, hi; } u;
  v16bf v;
};

__device__ __forceinline__ ushort_t f2bf(float f) {
  unsigned u = __builtin_bit_cast(unsigned, f);
  return (ushort_t)((u + 0x7FFFu + ((u >> 16) & 1u)) >> 16);  // RNE
}

__device__ __forceinline__ v8f wmma_bf16(v16bf a, v16bf b, v8f c) {
  // D = A(16x32 bf16) x B(32x16 bf16) + C(16x16 f32)
  return __builtin_amdgcn_wmma_f32_16x16x32_bf16(false, a, false, b, (short)0, c,
                                                 false, false);
}

// ---- WMMA A fragment loaders (16x32, 16-bit layout) ----
// lane<16: halfs 0-7 = K+0..7, halfs 8-15 = K+16..23 (M = lane)
// lane>=16: halfs 0-7 = K+8..15, halfs 8-15 = K+24..31 (M = lane-16)
// caller folds the per-lane +8 K offset (koff) into the pointer.
__device__ __forceinline__ v16bf load_a_f32(const float* __restrict__ p) {
  v16bf v;
#pragma unroll
  for (int i = 0; i < 8; ++i) v[i] = (bf16_t)p[i];
#pragma unroll
  for (int i = 0; i < 8; ++i) v[8 + i] = (bf16_t)p[16 + i];
  return v;
}

__device__ __forceinline__ v16bf load_a_u16(const ushort_t* __restrict__ p) {
  BFPack bp;
  bp.u.lo = *(const uint4*)p;          // K+0..7
  bp.u.hi = *(const uint4*)(p + 16);   // K+16..23
  return bp.v;
}

// ---- WMMA B fragment loader: pre-swizzled 1KB fragments in LDS ----
__device__ __forceinline__ v16bf load_b_lds(const char* smem, int frag, int lane) {
  const uint4* p = (const uint4*)(smem + frag * 1024 + lane * 32);
  BFPack bp;
  bp.u.lo = p[0];
  bp.u.hi = p[1];
  return bp.v;
}

// ---- Tensor Data Mover: 1D contiguous global -> LDS (data_size = 8B) ----
#if __has_builtin(__builtin_amdgcn_tensor_load_to_lds)
#define HAVE_TDM 1
__device__ __forceinline__ void tdm_load_1d(void* lds_dst, const void* gsrc,
                                            unsigned bytes) {
  unsigned long long ga = (unsigned long long)(size_t)gsrc;
  unsigned lds = (unsigned)(size_t)lds_dst;  // low 32 bits = LDS offset
  unsigned n8 = bytes >> 3;                  // elements of 8 bytes (<= 65535)
  v4u g0 = { 1u,                                            // count=1, user mode
             lds,                                           // lds_addr
             (unsigned)ga,                                  // global_addr[31:0]
             (unsigned)((ga >> 32) & 0x01FFFFFFu) | (2u << 30) }; // ga[56:32] | type=2
  v8i g1 = { (int)(3u << 16),                           // data_size = 8B
             (int)((n8 & 0xFFFFu) << 16),               // tensor_dim0[15:0]
             (int)(((n8 >> 16) & 0xFFFFu) | (1u << 16)),// dim0 hi | tensor_dim1=1
             (int)((n8 & 0xFFFFu) << 16),               // tile_dim0
             1,                                         // tile_dim1 = 1
             (int)n8,                                   // tensor_dim0_stride lo
             0, 0 };
  v4i gz = {0, 0, 0, 0};
#if defined(__clang_major__) && __clang_major__ >= 23
  v8i gz8 = {0, 0, 0, 0, 0, 0, 0, 0};
  __builtin_amdgcn_tensor_load_to_lds(g0, g1, gz, gz, gz8, 0);
#else
  __builtin_amdgcn_tensor_load_to_lds(g0, g1, gz, gz, 0);
#endif
}
#else
#define HAVE_TDM 0
#endif

// ---- device-wide barrier: cluster fast path + atomic fallback ----
__device__ __forceinline__ void grid_barrier(unsigned* syncv, int in_cluster) {
  __threadfence();
  __syncthreads();
  if (in_cluster) {
    if (threadIdx.x < 32) __builtin_amdgcn_s_cluster_barrier();
  } else if (threadIdx.x == 0) {
    unsigned* cnt = syncv;
    unsigned* gen = syncv + 16;
    unsigned g = __hip_atomic_load(gen, __ATOMIC_ACQUIRE, __HIP_MEMORY_SCOPE_AGENT);
    unsigned arrived =
        __hip_atomic_fetch_add(cnt, 1u, __ATOMIC_ACQ_REL, __HIP_MEMORY_SCOPE_AGENT);
    if (arrived == NWG - 1) {
      __hip_atomic_store(cnt, 0u, __ATOMIC_RELAXED, __HIP_MEMORY_SCOPE_AGENT);
      __hip_atomic_fetch_add(gen, 1u, __ATOMIC_ACQ_REL, __HIP_MEMORY_SCOPE_AGENT);
    } else {
      while (__hip_atomic_load(gen, __ATOMIC_ACQUIRE, __HIP_MEMORY_SCOPE_AGENT) == g)
        __builtin_amdgcn_s_sleep(2);
    }
  }
  __syncthreads();
  __threadfence();
}

// =====================  prep kernels  =====================

// Hidden weights -> bf16, pre-swizzled into B-fragment order.
// WB[wg][tile][ki][lane][h] ; K = ki*32 + (h>=8)*16 + (lane>=16)*8 + (h&7) ; N = lane&15
__global__ void pack_hidden_w(const float* __restrict__ Ws,
                              const float* __restrict__ Wa,
                              ushort_t* __restrict__ WB) {
  int tid = blockIdx.x * 256 + threadIdx.x;
  if (tid >= 16 * 4 * 40 * 512) return;
  int h    = tid & 15;
  int lane = (tid >> 4) & 31;
  int ki   = (tid >> 9) % 40;
  int rest = (tid >> 9) / 40;
  int tile = rest & 3;
  int wg   = rest >> 2;
  int k    = ki * 32 + ((h >> 3) << 4) + ((lane >> 4) << 3) + (h & 7);
  int nloc = (wg & 7) * 64 + tile * 16 + (lane & 15);
  float w = (wg < 8) ? Ws[(size_t)nloc * FANIN + k]
                     : Wa[(size_t)nloc * FANIN + k];
  WB[tid] = f2bf(w);
}

__global__ void pack_out_w(const float* __restrict__ Wo,
                           ushort_t* __restrict__ WOB) {
  int tid = blockIdx.x * 256 + threadIdx.x;
  if (tid >= 8 * 32 * 512) return;
  int h    = tid & 15;
  int lane = (tid >> 4) & 31;
  int ki   = (tid >> 9) & 31;
  int wg   = tid >> 14;
  int k    = ki * 32 + ((h >> 3) << 4) + ((lane >> 4) << 3) + (h & 7);
  int n    = wg * 16 + (lane & 15);
  WOB[tid] = f2bf(Wo[(size_t)n * HID + k]);
}

__global__ void zero_state(ushort_t* __restrict__ SB, unsigned* __restrict__ syncv) {
  int tid = blockIdx.x * 256 + threadIdx.x;
  if (tid < 2 * BATCH * HID) SB[tid] = 0;
  if (tid < 32) syncv[tid] = 0;
}

// =====================  persistent RSNN kernel  =====================

__global__ void __launch_bounds__(256, 1)
rsnn_persistent(const float* __restrict__ inp, const float* __restrict__ bs,
                const float* __restrict__ ba, const float* __restrict__ bo,
                const ushort_t* __restrict__ WB, const ushort_t* __restrict__ WOB,
                ushort_t* __restrict__ SB, unsigned* __restrict__ syncv,
                float* __restrict__ out) {
  extern __shared__ char smem[];  // [0,160K): 4 N-tiles x 40 k-frags; [160K,192K): 32 out frags
  const int wg   = blockIdx.x;     // 0..15
  const int tid  = threadIdx.x;    // 0..255 (8 waves)
  const int wave = tid >> 5;
  const int lane = tid & 31;
  const int col  = lane & 15;
  const int kh   = lane >> 4;      // lane-half (A/B K interleave select)
  const bool is_static = (wg < 8);
  const int m     = wave >> 1;       // M tile (batch rows m*16..)
  const int npair = wave & 1;
  const int nt0 = npair * 2, nt1 = nt0 + 1;
  const int nloc0 = (wg & 7) * 64 + nt0 * 16 + col;   // slice-local neuron index
  const int nloc1 = nloc0 + 16;
  const int scol0 = nloc0 + (is_static ? 0 : 512);    // spike-matrix column
  const int scol1 = nloc1 + (is_static ? 0 : 512);
  const float bias0 = is_static ? bs[nloc0] : ba[nloc0];
  const float bias1 = is_static ? bs[nloc1] : ba[nloc1];
  const int row  = m * 16 + col;   // batch row this lane loads A for
  const int koff = kh * 8;

  // ---- one-time weight staging into LDS ----
#if HAVE_TDM
  if (wave == 0) {
    tdm_load_1d(smem, WB + (size_t)wg * 81920, 163840);
    if (wg < 8) tdm_load_1d(smem + 163840, WOB + (size_t)wg * 16384, 32768);
    __builtin_amdgcn_s_wait_tensorcnt(0);
  }
#else
  {
    ushort_t* lw = (ushort_t*)smem;
    for (int i = tid; i < 81920; i += 256) lw[i] = WB[(size_t)wg * 81920 + i];
    if (wg < 8)
      for (int i = tid; i < 16384; i += 256) lw[81920 + i] = WOB[(size_t)wg * 16384 + i];
  }
#endif
  __syncthreads();

  const int in_cluster = (__builtin_amdgcn_cluster_id_x() != 0) ? 1 : 0;

  // persistent state (f32, in registers)
  v8f vs0 = {}, vs1 = {};   // membrane potential (vs or va)
  v8f ad0 = {}, ad1 = {};   // ALIF adaptation trace b
  v8f vo  = {};             // leaky-integrator readout
  const bool do_out = (wg < 8) && (wave < 4);
  const float biasO = do_out ? bo[wg * 16 + col] : 0.f;
  const int mo = wave;
  const int rowO = mo * 16 + col;

  for (int t = 0; t < T_STEPS; ++t) {
    const ushort_t* SBp = SB + (((t & 1) ^ 1) * (BATCH * HID));  // spikes[t-1]
    ushort_t*       SBc = SB + ((t & 1) * (BATCH * HID));        // spikes[t]
    const float* xin = inp + ((size_t)t * BATCH + row) * NIN;
    if (t + 1 < T_STEPS) __builtin_prefetch(xin + BATCH * NIN, 0, 1);

    // ---- hidden GEMM: [64,1280]x[1280,64-slice], K split input|recurrent ----
    v8f acc0 = {}, acc1 = {};
#pragma unroll 2
    for (int ki = 0; ki < 8; ++ki) {  // input part of x (f32 -> bf16 on the fly)
      v16bf a  = load_a_f32(xin + ki * 32 + koff);
      v16bf b0 = load_b_lds(smem, nt0 * 40 + ki, lane);
      v16bf b1 = load_b_lds(smem, nt1 * 40 + ki, lane);
      acc0 = wmma_bf16(a, b0, acc0);
      acc1 = wmma_bf16(a, b1, acc1);
    }
    const ushort_t* srow = SBp + row * HID + koff;
#pragma unroll 2
    for (int ki = 8; ki < 40; ++ki) {  // recurrent spikes (bf16)
      v16bf a  = load_a_u16(srow + (ki * 32 - 256));
      v16bf b0 = load_b_lds(smem, nt0 * 40 + ki, lane);
      v16bf b1 = load_b_lds(smem, nt1 * 40 + ki, lane);
      acc0 = wmma_bf16(a, b0, acc0);
      acc1 = wmma_bf16(a, b1, acc1);
    }

    // ---- neuron update + spike emission (C/D layout: M=r+8*kh, N=col) ----
    if (is_static) {
#pragma unroll
      for (int r = 0; r < 8; ++r) {
        int brow = m * 16 + r + 8 * kh;
        float v0 = DECAY * vs0[r] + acc0[r] + bias0;
        float s0 = (v0 >= THRESH) ? 1.f : 0.f;
        vs0[r] = v0 - s0 * THRESH;
        SBc[brow * HID + scol0] = (s0 != 0.f) ? (ushort_t)0x3F80 : (ushort_t)0;
        float v1 = DECAY * vs1[r] + acc1[r] + bias1;
        float s1 = (v1 >= THRESH) ? 1.f : 0.f;
        vs1[r] = v1 - s1 * THRESH;
        SBc[brow * HID + scol1] = (s1 != 0.f) ? (ushort_t)0x3F80 : (ushort_t)0;
      }
    } else {
#pragma unroll
      for (int r = 0; r < 8; ++r) {
        int brow = m * 16 + r + 8 * kh;
        float th0 = THRESH + BETA_ * ad0[r];
        float v0 = ALPHA_ * vs0[r] + acc0[r] + bias0;
        float s0 = (v0 >= th0) ? 1.f : 0.f;
        vs0[r] = v0 - s0 * th0;
        ad0[r] = RHO_ * ad0[r] + (1.f - RHO_) * s0;
        SBc[brow * HID + scol0] = (s0 != 0.f) ? (ushort_t)0x3F80 : (ushort_t)0;
        float th1 = THRESH + BETA_ * ad1[r];
        float v1 = ALPHA_ * vs1[r] + acc1[r] + bias1;
        float s1 = (v1 >= th1) ? 1.f : 0.f;
        vs1[r] = v1 - s1 * th1;
        ad1[r] = RHO_ * ad1[r] + (1.f - RHO_) * s1;
        SBc[brow * HID + scol1] = (s1 != 0.f) ? (ushort_t)0x3F80 : (ushort_t)0;
      }
    }

    grid_barrier(syncv, in_cluster);  // spikes[t] now device-visible

    // ---- readout: vo = KAPPA*vo + spikes[t] @ Wo.T + bo ----
    if (do_out) {
      v8f acc = {};
      const ushort_t* orow = SBc + rowO * HID + koff;
#pragma unroll 2
      for (int ki = 0; ki < 32; ++ki) {
        v16bf a = load_a_u16(orow + ki * 32);
        v16bf b = load_b_lds(smem, 160 + ki, lane);
        acc = wmma_bf16(a, b, acc);
      }
      vo = vo * KAPPA_ + acc + biasO;
    }
  }

  if (do_out) {
#pragma unroll
    for (int r = 0; r < 8; ++r) {
      int brow = mo * 16 + r + 8 * kh;
      out[brow * NOUT + wg * 16 + col] = vo[r];
    }
  }
}

// =====================  launcher  =====================

extern "C" void kernel_launch(void* const* d_in, const int* in_sizes, int n_in,
                              void* d_out, int out_size, void* d_ws, size_t ws_size,
                              hipStream_t stream) {
  const float* inp = (const float*)d_in[0];
  const float* Ws  = (const float*)d_in[1];
  const float* bs  = (const float*)d_in[2];
  const float* Wa  = (const float*)d_in[3];
  const float* ba  = (const float*)d_in[4];
  const float* Wo  = (const float*)d_in[5];
  const float* bo  = (const float*)d_in[6];

  char* ws = (char*)d_ws;
  unsigned* syncv = (unsigned*)ws;
  ushort_t* WB  = (ushort_t*)(ws + WS_WB_OFF);
  ushort_t* WOB = (ushort_t*)(ws + WS_WOB_OFF);
  ushort_t* SB  = (ushort_t*)(ws + WS_SB_OFF);

  pack_hidden_w<<<(16 * 4 * 40 * 512 + 255) / 256, 256, 0, stream>>>(Ws, Wa, WB);
  pack_out_w<<<(8 * 32 * 512 + 255) / 256, 256, 0, stream>>>(Wo, WOB);
  zero_state<<<512, 256, 0, stream>>>(SB, syncv);

  // 16 WGs (one per WGP; = one max-size workgroup cluster), 8 waves each,
  // 192KB dynamic LDS (hidden slice 160KB + readout slice 32KB) of the 320KB WGP pool.
  rsnn_persistent<<<16, 256, 196608, stream>>>(inp, bs, ba, bo, WB, WOB, SB, syncv,
                                               (float*)d_out);
  (void)in_sizes; (void)n_in; (void)out_size; (void)ws_size;
}
